// CRFLoss_74113955660354
// MI455X (gfx1250) — matrix operand order repaired and checked by reading, not compiled, not run on hardware
//
#include <hip/hip_runtime.h>
#include <hip/hip_bf16.h>
#include <math.h>

#define BS 512
#define SL 1024
#define NC 64

#define LOG2E 1.4426950408889634f
#define LN2   0.6931471805599453f

typedef __attribute__((ext_vector_type(16))) _Float16 v16h;
typedef __attribute__((ext_vector_type(8)))  _Float16 v8h;
typedef __attribute__((ext_vector_type(8)))  float    v8f;

__device__ __forceinline__ float hmax16(float x) {
#pragma unroll
  for (int m = 8; m >= 1; m >>= 1) x = fmaxf(x, __shfl_xor(x, m, 32));
  return x;
}
__device__ __forceinline__ float hsum16(float x) {
#pragma unroll
  for (int m = 8; m >= 1; m >>= 1) x += __shfl_xor(x, m, 32);
  return x;
}

// Raw hardware transcendentals (valid here: args are well inside normal range)
__device__ __forceinline__ float fast_exp(float x) {
  return __builtin_amdgcn_exp2f(x * LOG2E);      // v_mul + v_exp_f32
}
__device__ __forceinline__ float fast_log(float x) {
  return __builtin_amdgcn_logf(x) * LN2;          // v_log_f32 + v_mul
}

// Forward scan in the exp-domain: one wave32 per 16-row batch tile.
// alpha held in WMMA C-layout f32 accumulators; exp(T) pinned in B-layout
// f16 fragments; per-step C->A transpose through 2KB LDS.
// Single wave per block: LDS ops are in-order within a wave (DScnt), so a
// compiler-only wave_barrier replaces __syncthreads (no LOADcnt drain).
__global__ __launch_bounds__(32) void crf_scan_kernel(
    const float* __restrict__ inputs,   // [BS][SL][NC] f32
    const int*   __restrict__ masks,    // [BS][SL] int32 (0/1)
    const float* __restrict__ trans,    // [NC][NC] f32
    float*       __restrict__ log_norm) // [BS]
{
  __shared__ __align__(16) _Float16 ldsE[16 * NC];

  const int lane = threadIdx.x & 31;
  const int hi   = lane >> 4;   // half-wave
  const int lo   = lane & 15;
  const int b0   = blockIdx.x * 16;

  // --- expT in WMMA B layout (B is 32x16 per fragment): k = kt*32 + 16*hi + h, n = nt*16 + lo
  v16h Bf[4][2];
#pragma unroll
  for (int nt = 0; nt < 4; ++nt) {
    const int n = nt * 16 + lo;
#pragma unroll
    for (int kt = 0; kt < 2; ++kt) {
      v16h f;
#pragma unroll
      for (int h = 0; h < 16; ++h) {
        const int k = kt * 32 + hi * 16 + h;
        f[h] = (_Float16)fast_exp(trans[k * NC + n]);
      }
      Bf[nt][kt] = f;
    }
  }

  // --- alpha = inputs[:,0,:]  (C layout: b = b0 + v + 8*hi, j = nt*16 + lo)
  v8f acc[4];
#pragma unroll
  for (int nt = 0; nt < 4; ++nt)
#pragma unroll
    for (int v = 0; v < 8; ++v) {
      const int b = b0 + v + 8 * hi;
      acc[nt][v] = inputs[(size_t)b * SL * NC + nt * 16 + lo];
    }

  for (int t = 1; t < SL; ++t) {
    // 0) issue this step's mask + x_t loads FIRST; they stay outstanding
    //    across the max/exp/LDS/WMMA phases (no barrier drains LOADcnt).
    float xv[4][8];
    float wv[8];                         // 1.0 = keep old alpha, 0.0 = update
    const int tp = (t + 8 < SL) ? (t + 8) : (SL - 1);
#pragma unroll
    for (int v = 0; v < 8; ++v) {
      const int b = b0 + v + 8 * hi;
      wv[v] = (float)masks[(size_t)b * SL + t];
      const float* xr = inputs + ((size_t)b * SL + t) * NC;
#pragma unroll
      for (int nt = 0; nt < 4; ++nt) xv[nt][v] = xr[nt * 16 + lo];
      __builtin_prefetch(inputs + ((size_t)b * SL + tp) * NC + lane, 0, 3);
    }

    // 1) per-row max over the 64 columns (4 local values + 16-lane shuffle tree)
    float mb[8];
#pragma unroll
    for (int v = 0; v < 8; ++v) {
      float p = fmaxf(fmaxf(acc[0][v], acc[1][v]), fmaxf(acc[2][v], acc[3][v]));
      mb[v] = hmax16(p);
    }
    // 2) E = exp(alpha - rowmax) -> LDS, row-major f16 [16][64]
#pragma unroll
    for (int v = 0; v < 8; ++v) {
      const int row = v + 8 * hi;
#pragma unroll
      for (int nt = 0; nt < 4; ++nt)
        ldsE[row * NC + nt * 16 + lo] = (_Float16)fast_exp(acc[nt][v] - mb[v]);
    }
    __builtin_amdgcn_wave_barrier();   // scheduling fence only; DS is in-order
    // 3) A fragments (16x32 f16): m = lo; k = kt*32 + 16*hi + h
    //    -> two aligned 16B LDS loads per fragment
    v16h A[2];
#pragma unroll
    for (int kt = 0; kt < 2; ++kt) {
      const v8h g0 = *reinterpret_cast<const v8h*>(&ldsE[lo * NC + kt * 32 + hi * 16]);
      const v8h g1 = *reinterpret_cast<const v8h*>(&ldsE[lo * NC + kt * 32 + hi * 16 + 8]);
      v16h a;
#pragma unroll
      for (int h = 0; h < 8; ++h) { a[h] = g0[h]; a[h + 8] = g1[h]; }
      A[kt] = a;
    }
    // 4) S = E @ expT  (8 x v_wmma_f32_16x16x32_f16, f32 accumulate)
    v8f s[4];
#pragma unroll
    for (int nt = 0; nt < 4; ++nt) {
      v8f c = {};
      c = __builtin_amdgcn_wmma_f32_16x16x32_f16(false, A[0], false, Bf[nt][0],
                                                 (short)0, c, false, false);
      c = __builtin_amdgcn_wmma_f32_16x16x32_f16(false, A[1], false, Bf[nt][1],
                                                 (short)0, c, false, false);
      s[nt] = c;
    }
    // 5) branchless masked update: acc = na + w*(acc - na)
    //    (w in {0,1}; pure v_sub/v_fmac, cannot be if-converted to exec branches)
#pragma unroll
    for (int v = 0; v < 8; ++v) {
#pragma unroll
      for (int nt = 0; nt < 4; ++nt) {
        const float na = xv[nt][v] + mb[v] + fast_log(s[nt][v]);
        acc[nt][v] = fmaf(wv[v], acc[nt][v] - na, na);
      }
    }
    __builtin_amdgcn_wave_barrier();   // keep next iter's LDS stores behind loads
  }

  // --- log_norm[b] = logsumexp_j(alpha[b, :])
#pragma unroll
  for (int v = 0; v < 8; ++v) {
    float p = fmaxf(fmaxf(acc[0][v], acc[1][v]), fmaxf(acc[2][v], acc[3][v]));
    const float m = hmax16(p);
    float sum = 0.f;
#pragma unroll
    for (int nt = 0; nt < 4; ++nt) sum += fast_exp(acc[nt][v] - m);
    sum = hsum16(sum);
    if (lo == 0) log_norm[b0 + v + 8 * hi] = m + fast_log(sum);
  }
}

// unary + binary gold-path score, one block per batch row
__global__ __launch_bounds__(256) void crf_score_kernel(
    const float* __restrict__ inputs,
    const int*   __restrict__ masks,
    const int*   __restrict__ tags,
    const float* __restrict__ trans,
    float*       __restrict__ score)   // [BS]
{
  const int b = blockIdx.x;
  float s = 0.f;
  for (int t = threadIdx.x; t < SL; t += 256) {
    if (!masks[(size_t)b * SL + t]) {
      const int tag = tags[(size_t)b * SL + t];
      s += inputs[((size_t)b * SL + t) * NC + tag];
      if (t >= 1) s += trans[tags[(size_t)b * SL + t - 1] * NC + tag];
    }
  }
  __shared__ float red[256];
  red[threadIdx.x] = s;
  __syncthreads();
  for (int off = 128; off > 0; off >>= 1) {
    if (threadIdx.x < off) red[threadIdx.x] += red[threadIdx.x + off];
    __syncthreads();
  }
  if (threadIdx.x == 0) score[b] = red[0];
}

// loss = sum_b(log_norm - score) / BS  -> d_out[0]
__global__ __launch_bounds__(512) void crf_finalize_kernel(
    const float* __restrict__ log_norm,
    const float* __restrict__ score,
    float* __restrict__ out)
{
  __shared__ float red[512];
  const int tid = threadIdx.x;
  red[tid] = log_norm[tid] - score[tid];
  __syncthreads();
  for (int off = 256; off > 0; off >>= 1) {
    if (tid < off) red[tid] += red[tid + off];
    __syncthreads();
  }
  if (tid == 0) out[0] = red[0] / (float)BS;
}

// reference returns (loss, transitions): copy transitions to d_out[1..4096]
__global__ void crf_copy_trans_kernel(const float* __restrict__ trans,
                                      float* __restrict__ out)
{
  const int i = blockIdx.x * blockDim.x + threadIdx.x;
  if (i < NC * NC) out[1 + i] = trans[i];
}

extern "C" void kernel_launch(void* const* d_in, const int* in_sizes, int n_in,
                              void* d_out, int out_size, void* d_ws, size_t ws_size,
                              hipStream_t stream) {
  const float* inputs = (const float*)d_in[0];   // (512,1024,64) f32
  const int*   masks  = (const int*)d_in[1];     // (512,1024) bool -> int32
  const int*   tags   = (const int*)d_in[2];     // (512,1024) int
  const float* trans  = (const float*)d_in[3];   // (64,64) f32
  float* out = (float*)d_out;                    // [1 + 4096]
  float* ws  = (float*)d_ws;
  float* log_norm = ws;        // 512 floats
  float* score    = ws + BS;   // 512 floats

  crf_scan_kernel<<<BS / 16, 32, 0, stream>>>(inputs, masks, trans, log_norm);
  crf_score_kernel<<<BS, 256, 0, stream>>>(inputs, masks, tags, trans, score);
  crf_finalize_kernel<<<1, BS, 0, stream>>>(log_norm, score, out);
  crf_copy_trans_kernel<<<(NC * NC + 255) / 256, 256, 0, stream>>>(trans, out);
}
